// MultiHeadAttention_80290118631658
// MI455X (gfx1250) — matrix operand order repaired and checked by reading, not compiled, not run on hardware
//
#include <hip/hip_runtime.h>

typedef __bf16 bf16_t;
typedef __attribute__((ext_vector_type(16))) __bf16 v16bf;
typedef __attribute__((ext_vector_type(8)))  __bf16 v8bf;
typedef __attribute__((ext_vector_type(8)))  float  v8f;

#define D_  1024
#define H_  16
#define HD_ 64
#define B_  8
#define S_  1024

static __device__ __forceinline__ v8f wmma_bf16(v16bf a, v16bf b, v8f c) {
    return __builtin_amdgcn_wmma_f32_16x16x32_bf16(
        /*neg_a=*/false, a, /*neg_b=*/false, b,
        /*c_mod=*/(short)0, c, /*reuse_a=*/false, /*reuse_b=*/false);
}

static __device__ __forceinline__ v16bf cat8(v8bf lo, v8bf hi) {
    return __builtin_shufflevector(lo, hi, 0,1,2,3,4,5,6,7,8,9,10,11,12,13,14,15);
}

static __device__ __forceinline__ v8f zero8() {
    v8f z;
#pragma unroll
    for (int j = 0; j < 8; ++j) z[j] = 0.0f;
    return z;
}

// ---------------- fp32 -> bf16 convert ----------------
__global__ void mha_cvt_bf16(const float* __restrict__ src, bf16_t* __restrict__ dst, int n) {
    int i = blockIdx.x * blockDim.x + threadIdx.x;
    if (i < n) dst[i] = (bf16_t)src[i];
}

// W[k*N + n] (fp32, row-major KxN) -> Wt[n*K + k] (bf16, NxK)
__global__ void mha_transpose_cvt(const float* __restrict__ W, bf16_t* __restrict__ Wt, int K, int N) {
    int i = blockIdx.x * blockDim.x + threadIdx.x;
    if (i < K * N) {
        int n = i / K, k = i % K;
        Wt[i] = (bf16_t)W[(size_t)k * N + n];
    }
}

// ---------------- bf16 WMMA GEMM: out = A[MxK] * Wt[NxK]^T + bias ----------------
// Block = 4 waves; block tile 128(M) x 64(N); wave tile 32x64.
// B tile (64 x 32k) staged via async global->LDS, double-buffered (ASYNCcnt).
// mode 0/1: out bf16 at [B,H,S,HD]; mode 2: out bf16 at [B,H,HD,S]; mode 3: out fp32 [M,N]
__global__ __launch_bounds__(128) void mha_gemm_bf16(
    const bf16_t* __restrict__ A, const bf16_t* __restrict__ Wt,
    const float* __restrict__ bias, void* __restrict__ out,
    int M, int N, int K, int mode)
{
    __shared__ __align__(16) bf16_t btile[2][64][32];   // [buf][n][k], 2 x 4KB

    const int warp = threadIdx.x >> 5;
    const int lane = threadIdx.x & 31;
    const int hf   = lane >> 4;
    const int lx   = lane & 15;

    const int nblocks = N / 64;
    const int mt = blockIdx.x / nblocks;
    const int nb = blockIdx.x % nblocks;
    const int mwave = mt * 128 + warp * 32;

    const bf16_t* arow0 = A + (size_t)(mwave + lx) * K;
    const bf16_t* arow1 = A + (size_t)(mwave + 16 + lx) * K;

    const unsigned ldsbase = (unsigned)(uintptr_t)(&btile[0][0][0]);

    // async-stage B tile: 256 16B chunks, 64 per wave (2 wave-instructions)
    auto stage = [&](int buf, int kk) {
#pragma unroll
        for (int i = 0; i < 2; ++i) {
            const int chunk = warp * 64 + i * 32 + lane;  // 0..255
            const int n  = chunk >> 2;
            const int kc = (chunk & 3) * 8;
            const unsigned long long gaddr =
                (unsigned long long)(Wt + (size_t)(nb * 64 + n) * K + kk + kc);
            const unsigned laddr = ldsbase + (unsigned)(buf * 4096 + n * 64 + kc * 2);
            asm volatile("global_load_async_to_lds_b128 %0, %1, off"
                         :: "v"(laddr), "v"(gaddr) : "memory");
        }
    };

    v8f acc[2][4];
#pragma unroll
    for (int ms = 0; ms < 2; ++ms)
#pragma unroll
        for (int n = 0; n < 4; ++n) acc[ms][n] = zero8();

    const int nk = K / 32;
    stage(0, 0);
    for (int t = 0; t < nk; ++t) {
        const int cur = t & 1;
        const int kk  = t * 32;
        if (t + 1 < nk) {
            stage(cur ^ 1, kk + 32);
            asm volatile("s_wait_asynccnt 0x2" ::: "memory");  // current buf resident
        } else {
            asm volatile("s_wait_asynccnt 0x0" ::: "memory");
        }
        __syncthreads();   // all waves' async copies visible

        v16bf a0 = cat8(*(const v8bf*)(arow0 + kk + hf * 8),
                        *(const v8bf*)(arow0 + kk + 16 + hf * 8));
        v16bf a1 = cat8(*(const v8bf*)(arow1 + kk + hf * 8),
                        *(const v8bf*)(arow1 + kk + 16 + hf * 8));
#pragma unroll
        for (int n = 0; n < 4; ++n) {
            // B fragment 32x16: b[j] = W[k = kk + hf*8 + j (j<8), kk + 16 + hf*8 + j-8 (j>=8)][n]
            v16bf bfrag = cat8(*(const v8bf*)(&btile[cur][n * 16 + lx][hf * 8]),
                               *(const v8bf*)(&btile[cur][n * 16 + lx][16 + hf * 8]));
            acc[0][n] = wmma_bf16(a0, bfrag, acc[0][n]);
            acc[1][n] = wmma_bf16(a1, bfrag, acc[1][n]);
        }
        __syncthreads();   // done reading before buffer is re-targeted
    }

#pragma unroll
    for (int ms = 0; ms < 2; ++ms) {
#pragma unroll
        for (int n = 0; n < 4; ++n) {
            const int col = nb * 64 + n * 16 + lx;
            const float bv = bias[col];
#pragma unroll
            for (int r = 0; r < 8; ++r) {
                const int row = mwave + ms * 16 + r + hf * 8;
                const float v = acc[ms][n][r] + bv;
                if (mode == 3) {
                    ((float*)out)[(size_t)row * N + col] = v;
                } else {
                    const int bidx = row / S_, s = row % S_;
                    const int h = col / HD_, hd = col % HD_;
                    size_t idx;
                    if (mode == 2) idx = (((size_t)bidx * H_ + h) * HD_ + hd) * S_ + s;
                    else           idx = (((size_t)bidx * H_ + h) * S_  + s ) * HD_ + hd;
                    ((bf16_t*)out)[idx] = (bf16_t)v;
                }
            }
        }
    }
}

// ---------------- flash attention: one wave per 16 query rows ----------------
__global__ __launch_bounds__(128) void mha_attn(
    const bf16_t* __restrict__ Q,    // [B,H,S,HD]
    const bf16_t* __restrict__ Kt,   // [B,H,S,HD]
    const bf16_t* __restrict__ Vt,   // [B,H,HD,S]
    const bf16_t* __restrict__ relb, // [S,HD] bf16
    const int*    __restrict__ amask,// [B,S]
    bf16_t* __restrict__ ctx)        // [B,S,D] with col = h*HD+hd
{
    __shared__ __align__(32) bf16_t plds[4][16][32];

    const int warp = threadIdx.x >> 5;
    const int lane = threadIdx.x & 31;
    const int hf   = lane >> 4;
    const int lx   = lane & 15;

    const int wg = blockIdx.x * 4 + warp;   // 0 .. B*H*(S/16)-1
    const int bh = wg >> 6;                 // / (S/16)
    const int qt = wg & 63;
    const int b  = bh >> 4;                 // / H
    const int h  = bh & 15;

    const bf16_t* qbase = Q  + ((size_t)bh * S_ + qt * 16) * HD_;
    const bf16_t* kbase = Kt + (size_t)bh * S_ * HD_;
    const bf16_t* vbase = Vt + (size_t)bh * HD_ * S_;
    const int*    mrow  = amask + b * S_;

    v16bf qa[2];
    {
        const bf16_t* qrow = qbase + lx * HD_;
#pragma unroll
        for (int ks = 0; ks < 2; ++ks) {
            v8bf lo = *(const v8bf*)(qrow + ks * 32 + hf * 16);
            v8bf hi = *(const v8bf*)(qrow + ks * 32 + hf * 16 + 8);
            qa[ks] = cat8(lo, hi);
        }
    }

    const float scale = 0.125f;   // HD^-0.5
    float mrun[8], lrun[8];
    v8f cacc[4];
#pragma unroll
    for (int r = 0; r < 8; ++r) { mrun[r] = -1e30f; lrun[r] = 0.0f; }
#pragma unroll
    for (int n = 0; n < 4; ++n) cacc[n] = zero8();

    for (int tc = 0; tc < S_ / 32; ++tc) {
        const int t0 = tc * 32;
        v8f sqk0 = zero8(), sqk1 = zero8(), sr0 = zero8(), sr1 = zero8();
#pragma unroll
        for (int ks = 0; ks < 2; ++ks) {
            v16bf kb0 = *(const v16bf*)(kbase + (size_t)(t0 + lx)      * HD_ + ks * 32 + hf * 16);
            v16bf kb1 = *(const v16bf*)(kbase + (size_t)(t0 + 16 + lx) * HD_ + ks * 32 + hf * 16);
            v16bf rb0 = *(const v16bf*)(relb  + (size_t)(t0 + lx)      * HD_ + ks * 32 + hf * 16);
            v16bf rb1 = *(const v16bf*)(relb  + (size_t)(t0 + 16 + lx) * HD_ + ks * 32 + hf * 16);
            sqk0 = wmma_bf16(qa[ks], kb0, sqk0);
            sqk1 = wmma_bf16(qa[ks], kb1, sqk1);
            sr0  = wmma_bf16(qa[ks], rb0, sr0);
            sr1  = wmma_bf16(qa[ks], rb1, sr1);
        }

        const bool valid0 = mrow[t0 + lx] != 0;
        const bool valid1 = mrow[t0 + 16 + lx] != 0;

        float s0[8], s1[8], p0[8], p1[8], mnew[8];
#pragma unroll
        for (int r = 0; r < 8; ++r) {
            s0[r] = sqk0[r] * scale + sr0[r];
            s1[r] = sqk1[r] * scale + sr1[r];
        }
#pragma unroll
        for (int r = 0; r < 8; ++r) {
            float mv = fmaxf(valid0 ? s0[r] : -1e30f, valid1 ? s1[r] : -1e30f);
#pragma unroll
            for (int off = 8; off >= 1; off >>= 1)
                mv = fmaxf(mv, __shfl_xor(mv, off, 16));
            mnew[r] = fmaxf(mrun[r], mv);
        }
#pragma unroll
        for (int r = 0; r < 8; ++r) {
            const float alpha = __expf(mrun[r] - mnew[r]);
            p0[r] = valid0 ? __expf(s0[r] - mnew[r]) : 0.0f;
            p1[r] = valid1 ? __expf(s1[r] - mnew[r]) : 0.0f;
            float rs = p0[r] + p1[r];
#pragma unroll
            for (int off = 8; off >= 1; off >>= 1)
                rs += __shfl_xor(rs, off, 16);
            lrun[r] = lrun[r] * alpha + rs;
            mrun[r] = mnew[r];
#pragma unroll
            for (int n = 0; n < 4; ++n) cacc[n][r] *= alpha;
        }
#pragma unroll
        for (int r = 0; r < 8; ++r) {
            plds[warp][r + hf * 8][lx]      = (bf16_t)p0[r];
            plds[warp][r + hf * 8][16 + lx] = (bf16_t)p1[r];
        }
        __syncthreads();
        v16bf pa;
        {
            const v8bf* rp = (const v8bf*)(&plds[warp][lx][0]);
            pa = cat8(rp[2 * hf], rp[2 * hf + 1]);
        }
        __syncthreads();
#pragma unroll
        for (int n = 0; n < 4; ++n) {
            v16bf vb = *(const v16bf*)(vbase + (size_t)(n * 16 + lx) * S_ + t0 + hf * 16);
            cacc[n] = wmma_bf16(pa, vb, cacc[n]);
        }
    }

#pragma unroll
    for (int r = 0; r < 8; ++r) {
        const int srow = qt * 16 + r + hf * 8;
        const float inv = 1.0f / fmaxf(lrun[r], 1e-30f);
#pragma unroll
        for (int n = 0; n < 4; ++n) {
            const int dcol = h * HD_ + n * 16 + lx;
            ctx[((size_t)b * S_ + srow) * D_ + dcol] = (bf16_t)(cacc[n][r] * inv);
        }
    }
}

// ---------------- host launcher ----------------
extern "C" void kernel_launch(void* const* d_in, const int* in_sizes, int n_in,
                              void* d_out, int out_size, void* d_ws, size_t ws_size,
                              hipStream_t stream) {
    const float* x    = (const float*)d_in[0];
    const float* rel  = (const float*)d_in[1];
    const int*   msk  = (const int*)d_in[2];
    const float* Wq   = (const float*)d_in[3];
    const float* bq   = (const float*)d_in[4];
    const float* Wk   = (const float*)d_in[5];
    const float* bk   = (const float*)d_in[6];
    const float* Wv   = (const float*)d_in[7];
    const float* bv   = (const float*)d_in[8];
    const float* Wo   = (const float*)d_in[9];
    const float* bo   = (const float*)d_in[10];
    float* out = (float*)d_out;

    const size_t NX = (size_t)B_ * S_ * D_;   // 8388608
    const size_t NW = (size_t)D_ * D_;        // 1048576
    const size_t NR = (size_t)S_ * HD_;       // 65536

    char* ws = (char*)d_ws;
    size_t off = 0;
    auto take = [&](size_t elems) -> bf16_t* {
        bf16_t* p = (bf16_t*)(ws + off);
        off += ((elems * sizeof(bf16_t) + 255) & ~(size_t)255);
        return p;
    };
    bf16_t* x_bf   = take(NX);   // reused as ctx after the V projection
    bf16_t* rel_bf = take(NR);
    bf16_t* Wtq    = take(NW);
    bf16_t* Wtk    = take(NW);
    bf16_t* Wtv    = take(NW);
    bf16_t* Wto    = take(NW);
    bf16_t* Qb     = take(NX);
    bf16_t* Kb     = take(NX);
    bf16_t* Vtb    = take(NX);

    mha_cvt_bf16<<<(int)(NX / 256), 256, 0, stream>>>(x, x_bf, (int)NX);
    mha_cvt_bf16<<<(int)(NR / 256), 256, 0, stream>>>(rel, rel_bf, (int)NR);
    mha_transpose_cvt<<<(int)(NW / 256), 256, 0, stream>>>(Wq, Wtq, D_, D_);
    mha_transpose_cvt<<<(int)(NW / 256), 256, 0, stream>>>(Wk, Wtk, D_, D_);
    mha_transpose_cvt<<<(int)(NW / 256), 256, 0, stream>>>(Wv, Wtv, D_, D_);
    mha_transpose_cvt<<<(int)(NW / 256), 256, 0, stream>>>(Wo, Wto, D_, D_);

    // GEMMs: block tile 128x64 -> (8192/128)*(1024/64) = 1024 blocks of 128 threads
    const int M = B_ * S_;
    const int gemm_blocks = (M / 128) * (D_ / 64);
    mha_gemm_bf16<<<gemm_blocks, 128, 0, stream>>>(x_bf, Wtq, bq, Qb,  M, D_, D_, 0);
    mha_gemm_bf16<<<gemm_blocks, 128, 0, stream>>>(x_bf, Wtk, bk, Kb,  M, D_, D_, 1);
    mha_gemm_bf16<<<gemm_blocks, 128, 0, stream>>>(x_bf, Wtv, bv, Vtb, M, D_, D_, 2);

    const int attn_blocks = B_ * H_ * (S_ / 16) / 4;    // 2048
    mha_attn<<<attn_blocks, 128, 0, stream>>>(Qb, Kb, Vtb, rel_bf, msk, x_bf /*ctx*/);

    mha_gemm_bf16<<<gemm_blocks, 128, 0, stream>>>(x_bf, Wto, bo, out, M, D_, D_, 3);
}
